// SelfAttention1DLayer_39642548142321
// MI455X (gfx1250) — matrix operand order repaired and checked by reading, not compiled, not run on hardware
//
#include <hip/hip_runtime.h>

// ---------------------------------------------------------------------------
// SelfAttention1DLayer for MI455X (gfx1250, wave32, WMMA)
//   H     = tanh(x @ wk^T)         (B*T, DA)  bf16   v_wmma_f32_16x16x32_bf16
//   sim   = kernel @ H_b^T         (R, T)     fp32 accum
//   sm    = softmax_t(sim)         (R, T)     bf16
//   out_b = sm @ x_b               (R, D)     fp32
// GEMMs: 64x64 macro-tile per 128-thread (4-wave) workgroup, A/B staged in
// LDS (double-buffered, async global->LDS), each wave owns a 32x32 quadrant
// = 2x2 WMMA subtiles.
// ---------------------------------------------------------------------------

typedef __bf16 bf16_t;
typedef __attribute__((ext_vector_type(16))) __bf16 v16bf;
typedef __attribute__((ext_vector_type(8)))  __bf16 v8bf;
typedef __attribute__((ext_vector_type(8)))  float  v8f;
typedef int v4i_t __attribute__((vector_size(16)));

#define BB  8
#define TT  2048
#define DD  1024
#define RR  2048
#define DAA 512

#define LDS_PAD 8           // row stride 40 bf16 = 80B: conflict-free strides
#define LDS_LD  (32 + LDS_PAD)

#if defined(__has_builtin)
#if __has_builtin(__builtin_amdgcn_global_load_async_to_lds_b128) && \
    __has_builtin(__builtin_amdgcn_s_wait_asynccnt)
#define USE_ASYNC 1
#endif
#endif
#ifndef USE_ASYNC
#define USE_ASYNC 0
#endif

// ---- float -> bf16 (round to nearest even) --------------------------------
__device__ __forceinline__ bf16_t f2bf(float f) {
  unsigned u = __builtin_bit_cast(unsigned, f);
  u += 0x7FFFu + ((u >> 16) & 1u);
  unsigned short h = (unsigned short)(u >> 16);
  return __builtin_bit_cast(bf16_t, h);
}

// ---- 16B global -> LDS copy (async, ASYNCcnt-tracked) ---------------------
__device__ __forceinline__ void stage16(const bf16_t* __restrict__ g, bf16_t* l) {
#if USE_ASYNC
  __builtin_amdgcn_global_load_async_to_lds_b128(
      (__attribute__((address_space(1))) v4i_t*)(void*)const_cast<bf16_t*>(g),
      (__attribute__((address_space(3))) v4i_t*)(void*)l, 0, 0);
#else
  *(v8bf*)l = *(const v8bf*)g;
#endif
}

__device__ __forceinline__ void wait_stage() {
#if USE_ASYNC
  __builtin_amdgcn_s_wait_asynccnt(0);
#endif
}

// ---- WMMA fragment loads from LDS tile (ISA 7.12.2 layouts, wave32) -------
// A fragment 16x32 (MxK): lanes 0-15 row=lane, K 0-7 & 16-23; lanes 16-31
// row=lane-16, K 8-15 & 24-31.
__device__ __forceinline__ v16bf lds_a_frag(const bf16_t* base, int lane) {
  const bf16_t* p = base + (size_t)(lane & 15) * LDS_LD + ((lane >> 4) << 3);
  union { v16bf v; v8bf h[2]; } u;
  u.h[0] = *(const v8bf*)p;
  u.h[1] = *(const v8bf*)(p + 16);
  return u.v;
}
// B fragment 32x16 (KxN) from N-major tile: lanes 0-15 col=lane K 0-7 & 16-23;
// lanes 16-31 col=lane-16 K 8-15 & 24-31.
__device__ __forceinline__ v16bf lds_b_frag(const bf16_t* base, int lane) {
  const bf16_t* p = base + (size_t)(lane & 15) * LDS_LD + ((lane >> 4) << 3);
  union { v16bf v; v8bf h[2]; } u;
  u.h[0] = *(const v8bf*)p;
  u.h[1] = *(const v8bf*)(p + 16);
  return u.v;
}

__device__ __forceinline__ v8f wmma_bf16(v16bf a, v16bf b, v8f c) {
  // (neg_a, A, neg_b, B, c_mod, C, reuse_a, reuse_b)
  return __builtin_amdgcn_wmma_f32_16x16x32_bf16(false, a, false, b,
                                                 (short)0, c, false, false);
}

struct Tile32 { v8f c[2][2]; };

// ---- 64x64 macro-tile GEMM: C(MxN) = A(MxK,row-major) @ Bt(NxK,row-major)^T
// TANH_BF16: apply tanhf, store bf16; else store fp32.
template <bool TANH_BF16>
__global__ __launch_bounds__(128) void gemm64(const bf16_t* __restrict__ A, int lda,
                                              const bf16_t* __restrict__ Bt, int ldb,
                                              int K, void* __restrict__ Dst, int ldd) {
  __shared__ bf16_t As[2][64][LDS_LD];
  __shared__ bf16_t Bs[2][64][LDS_LD];

  const int tid  = threadIdx.x;
  const int lane = tid & 31;
  const int wave = tid >> 5;
  const int rowB = blockIdx.y * 64;
  const int colB = blockIdx.x * 64;
  const int rowQ = (wave >> 1) * 32;   // wave's quadrant inside macro-tile
  const int colQ = (wave & 1) * 32;

  // staging: thread copies one 16-element span of one A row and one Bt row
  const int srow = tid >> 1;           // 0..63
  const int scol = (tid & 1) * 16;     // 0 or 16
  const bf16_t* gA = A  + (size_t)(rowB + srow) * lda + scol;
  const bf16_t* gB = Bt + (size_t)(colB + srow) * ldb + scol;

  auto issue = [&](int buf, int k) {
    stage16(gA + k,     &As[buf][srow][scol]);
    stage16(gA + k + 8, &As[buf][srow][scol + 8]);
    stage16(gB + k,     &Bs[buf][srow][scol]);
    stage16(gB + k + 8, &Bs[buf][srow][scol + 8]);
  };

  Tile32 acc = {};
  const int nk = K >> 5;
  issue(0, 0);
  for (int i = 0; i < nk; ++i) {
    wait_stage();
    __syncthreads();                       // buf i fully staged by all waves
    if (i + 1 < nk) issue((i + 1) & 1, (i + 1) << 5);
    const int buf = i & 1;
    v16bf a0 = lds_a_frag(&As[buf][rowQ][0],      lane);
    v16bf a1 = lds_a_frag(&As[buf][rowQ + 16][0], lane);
    v16bf b0 = lds_b_frag(&Bs[buf][colQ][0],      lane);
    v16bf b1 = lds_b_frag(&Bs[buf][colQ + 16][0], lane);
    acc.c[0][0] = wmma_bf16(a0, b0, acc.c[0][0]);
    acc.c[0][1] = wmma_bf16(a0, b1, acc.c[0][1]);
    acc.c[1][0] = wmma_bf16(a1, b0, acc.c[1][0]);
    acc.c[1][1] = wmma_bf16(a1, b1, acc.c[1][1]);
    __syncthreads();                       // done reading buf i before overwrite
  }

  // C/D layout: lane->col = (lane&15); VGPR i -> row = i + (lane>=16 ? 8 : 0)
  const int col   = colB + colQ + (lane & 15);
  const int rbase = rowB + rowQ + ((lane >> 4) << 3);
  for (int mi = 0; mi < 2; ++mi)
    for (int ni = 0; ni < 2; ++ni)
      for (int i = 0; i < 8; ++i) {
        size_t idx = (size_t)(rbase + mi * 16 + i) * ldd + (col + ni * 16);
        if (TANH_BF16)
          ((bf16_t*)Dst)[idx] = f2bf(tanhf(acc.c[mi][ni][i]));
        else
          ((float*)Dst)[idx] = acc.c[mi][ni][i];
      }
}

// ---- conversions ----------------------------------------------------------
__global__ void cvt_f32_to_bf16(const float* __restrict__ in,
                                bf16_t* __restrict__ out, int n) {
  int i = blockIdx.x * blockDim.x + threadIdx.x;
  int stride = gridDim.x * blockDim.x;
  for (; i < n; i += stride) out[i] = f2bf(in[i]);
}

// xT[b][d][t] = bf16(x[b][t][d]); grid (T/32, D/32, B), block (32, 8)
__global__ void transpose_x_bf16(const float* __restrict__ x,
                                 bf16_t* __restrict__ xT) {
  __shared__ bf16_t tile[32][33];
  int b  = blockIdx.z;
  int t0 = blockIdx.x * 32, d0 = blockIdx.y * 32;
  int lt = threadIdx.x, ly = threadIdx.y;
  for (int j = 0; j < 32; j += 8)
    tile[ly + j][lt] = f2bf(x[((size_t)b * TT + (t0 + ly + j)) * DD + (d0 + lt)]);
  __syncthreads();
  for (int j = 0; j < 32; j += 8)
    xT[((size_t)b * DD + (d0 + ly + j)) * TT + (t0 + lt)] = tile[lt][ly + j];
}

// ---- softmax over t per (r): block = 256, grid = R ------------------------
__global__ __launch_bounds__(256) void softmax_rows(const float* __restrict__ sim,
                                                    bf16_t* __restrict__ sm) {
  __shared__ float red[256];
  int r = blockIdx.x;
  const float* row = sim + (size_t)r * TT;
  float v[TT / 256];
  float mx = -3.402823466e38f;
  for (int i = 0; i < TT / 256; ++i) {
    v[i] = row[threadIdx.x + i * 256];
    mx = fmaxf(mx, v[i]);
  }
  red[threadIdx.x] = mx; __syncthreads();
  for (int off = 128; off > 0; off >>= 1) {
    if (threadIdx.x < off)
      red[threadIdx.x] = fmaxf(red[threadIdx.x], red[threadIdx.x + off]);
    __syncthreads();
  }
  mx = red[0]; __syncthreads();
  float sum = 0.f;
  for (int i = 0; i < TT / 256; ++i) { v[i] = __expf(v[i] - mx); sum += v[i]; }
  red[threadIdx.x] = sum; __syncthreads();
  for (int off = 128; off > 0; off >>= 1) {
    if (threadIdx.x < off) red[threadIdx.x] += red[threadIdx.x + off];
    __syncthreads();
  }
  float inv = 1.0f / red[0];
  for (int i = 0; i < TT / 256; ++i)
    sm[(size_t)r * TT + threadIdx.x + i * 256] = f2bf(v[i] * inv);
}

// ---------------------------------------------------------------------------
extern "C" void kernel_launch(void* const* d_in, const int* in_sizes, int n_in,
                              void* d_out, int out_size, void* d_ws, size_t ws_size,
                              hipStream_t stream) {
  (void)in_sizes; (void)n_in; (void)out_size; (void)ws_size;
  const float* x      = (const float*)d_in[0];   // (B, T, D)
  const float* kernel = (const float*)d_in[1];   // (R, DA)
  const float* wk     = (const float*)d_in[2];   // (DA, D)
  float* out = (float*)d_out;                    // (B, R, D)

  char* ws = (char*)d_ws;
  size_t off = 0;
  auto carve = [&](size_t bytes) { size_t p = off; off += (bytes + 255) & ~(size_t)255; return p; };
  bf16_t* x_bf  = (bf16_t*)(ws + carve((size_t)BB * TT * DD * 2));   // 32 MB
  bf16_t* xT_bf = (bf16_t*)(ws + carve((size_t)BB * DD * TT * 2));   // 32 MB
  bf16_t* wk_bf = (bf16_t*)(ws + carve((size_t)DAA * DD * 2));       //  1 MB
  bf16_t* kr_bf = (bf16_t*)(ws + carve((size_t)RR * DAA * 2));       //  2 MB
  bf16_t* H_bf  = (bf16_t*)(ws + carve((size_t)BB * TT * DAA * 2));  // 16 MB
  float*  sim   = (float*) (ws + carve((size_t)RR * TT * 4));        // 16 MB (per batch)
  bf16_t* sm    = (bf16_t*)(ws + carve((size_t)RR * TT * 2));        //  8 MB (per batch)

  // Stage 0: precision conversion + x transpose
  cvt_f32_to_bf16<<<dim3(4096), dim3(256), 0, stream>>>(x, x_bf, BB * TT * DD);
  cvt_f32_to_bf16<<<dim3(512),  dim3(256), 0, stream>>>(wk, wk_bf, DAA * DD);
  cvt_f32_to_bf16<<<dim3(1024), dim3(256), 0, stream>>>(kernel, kr_bf, RR * DAA);
  transpose_x_bf16<<<dim3(TT / 32, DD / 32, BB), dim3(32, 8), 0, stream>>>(x, xT_bf);

  // Stage 1: H = tanh(X @ Wk^T), all batches as one (B*T x D) GEMM
  gemm64<true><<<dim3(DAA / 64, (BB * TT) / 64), dim3(128), 0, stream>>>(
      x_bf, DD, wk_bf, DD, DD, (void*)H_bf, DAA);

  // Stage 2/3/4 per batch (sim/sm buffers reused; stream-ordered)
  for (int b = 0; b < BB; ++b) {
    const bf16_t* Hb  = H_bf  + (size_t)b * TT * DAA;
    const bf16_t* XTb = xT_bf + (size_t)b * DD * TT;
    float*        Ob  = out   + (size_t)b * RR * DD;
    gemm64<false><<<dim3(TT / 64, RR / 64), dim3(128), 0, stream>>>(
        kr_bf, DAA, Hb, DAA, DAA, (void*)sim, TT);
    softmax_rows<<<dim3(RR), dim3(256), 0, stream>>>(sim, sm);
    gemm64<false><<<dim3(DD / 64, RR / 64), dim3(128), 0, stream>>>(
        sm, TT, XTb, TT, TT, (void*)Ob, DD);
  }
}
